// Ponita_19602230739217
// MI455X (gfx1250) — matrix operand order repaired and checked by reading, not compiled, not run on hardware
//
#include <hip/hip_runtime.h>
#include <math.h>

#define B_  2
#define N_  128
#define IN_ 16
#define H_  128
#define BD_ 32
#define O_  12
#define L_  2
#define WF_ 4

typedef __attribute__((ext_vector_type(16))) __bf16 v16bf;
typedef __attribute__((ext_vector_type(8)))  float  v8f;

// workspace layout (floats)
constexpr long OFF_ORI  = 0;                              // 36
constexpr long OFF_FB   = 64;                             // O*O*BD = 4608
constexpr long OFF_ROT  = OFF_FB  + (long)O_*O_*BD_;      // L*O*O*H = 36864
constexpr long OFF_H    = OFF_ROT + (long)L_*O_*O_*H_;    // B*N*O*H = 393216
constexpr long OFF_CONV = OFF_H   + (long)B_*N_*O_*H_;    // B*N*O*H = 393216
constexpr long OFF_KB   = OFF_CONV+ (long)B_*N_*O_*H_;    // B*N*N*O*BD = 12582912

__device__ __forceinline__ float gelu_t(float x) {
  float x3 = x * x * x;
  return 0.5f * x * (1.0f + tanhf(0.7978845608028654f * (x + 0.044715f * x3)));
}

// C(16x16,f32) += A(16xK,row-major,lda) * B(Kx16,row-major,ldb); K % 32 == 0.
// Fragments gathered per-lane from generic pointers (LDS or global).
__device__ __forceinline__ v8f wmma_rm(const float* A, int lda,
                                       const float* B, int ldb, int K, v8f c) {
  const int lane = threadIdx.x & 31;
  const int half = lane >> 4;
  const int mn   = lane & 15;
#pragma unroll
  for (int k0 = 0; k0 < K; k0 += 32) {
    v16bf a, b;
#pragma unroll
    for (int e = 0; e < 16; ++e) {
      int ka = k0 + e + half * 8 + ((e >= 8) ? 8 : 0);   // A 16-bit layout
      a[e] = (__bf16)A[mn * lda + ka];
      int kb = k0 + e + half * 8 + ((e >= 8) ? 8 : 0);   // B 16-bit layout (same k order as A)
      b[e] = (__bf16)B[kb * ldb + mn];
    }
    c = __builtin_amdgcn_wmma_f32_16x16x32_bf16(false, a, false, b,
                                                (short)0, c, false, false);
  }
  return c;
}

// ---------------- constants: fibonacci lattice, fiber basis, rot kernels ----
__global__ void __launch_bounds__(128)
k_precompute(const float* W1r, const float* b1r, const float* W2r,
             const float* b2r, const float* Wrot, float* ws) {
  float* ori = ws + OFF_ORI;
  float* fb  = ws + OFF_FB;
  float* rot = ws + OFF_ROT;
  const int tid = threadIdx.x;
  const float PI = 3.14159265358979323846f;
  if (tid < O_) {
    float i = (float)tid;
    float theta = fmodf(PI * i * (1.0f + sqrtf(5.0f)), 2.0f * PI);
    float phi = acosf(1.0f - 2.0f * (i + 0.5f) / (float)(O_ - 1 + 1));
    ori[tid * 3 + 0] = sinf(phi) * cosf(theta);
    ori[tid * 3 + 1] = sinf(phi) * sinf(theta);
    ori[tid * 3 + 2] = cosf(phi);
  }
  __syncthreads();
  __shared__ float sh1[H_];
  for (int r = 0; r < O_ * O_; ++r) {
    int p = r / O_, o = r % O_;
    float t = ori[p*3]*ori[o*3] + ori[p*3+1]*ori[o*3+1] + ori[p*3+2]*ori[o*3+2];
    float f0 = t, f1 = t * t, f2 = t * t * t;
    {
      int c = tid;
      float acc = b1r[c] + f0*W1r[0*H_+c] + f1*W1r[1*H_+c] + f2*W1r[2*H_+c];
      sh1[c] = gelu_t(acc);
    }
    __syncthreads();
    if (tid < BD_) {
      float acc = b2r[tid];
      for (int c = 0; c < H_; ++c) acc += sh1[c] * W2r[c * BD_ + tid];
      fb[r * BD_ + tid] = gelu_t(acc);
    }
    __syncthreads();
  }
  int c = tid;
  for (int l = 0; l < L_; ++l)
    for (int p = 0; p < O_; ++p)
      for (int o = 0; o < O_; ++o) {
        float acc = 0.f;
        const float* fbr = fb + (p * O_ + o) * BD_;
        const float* w = Wrot + (long)l * BD_ * H_;
        for (int d = 0; d < BD_; ++d) acc += fbr[d] * w[d * H_ + c];
        rot[(((long)l * O_ + p) * O_ + o) * H_ + c] = acc;
      }
}

// ---------------- h0 = x @ We, broadcast over O ----------------------------
__global__ void __launch_bounds__(128)
k_embed(const float* x, const float* We, float* ws) {
  int bn = blockIdx.x;
  int c = threadIdx.x;
  const float* xr = x + bn * IN_;
  float acc = 0.f;
  for (int k = 0; k < IN_; ++k) acc += xr[k] * We[k * H_ + c];
  float* h = ws + OFF_H + (long)bn * O_ * H_;
  for (int o = 0; o < O_; ++o) h[o * H_ + c] = acc;
}

// ---------------- kernel_basis: invariants -> poly(14) -> 128 -> 32 --------
__global__ void __launch_bounds__(128)
k_basis(const float* pos, const float* W1s, const float* b1s,
        const float* W2s, const float* b2s, float* ws) {
  __shared__ float sW1[32][H_];    // padded K 14->32
  __shared__ float sW2[H_][BD_];
  __shared__ float sb1[H_];
  __shared__ float sb2[BD_];
  __shared__ float sF[64][32];     // poly features, zero-padded
  __shared__ float sH1[64][H_];
  const float* ori = ws + OFF_ORI;
  float* kb = ws + OFF_KB;
  const int tid = threadIdx.x;

  for (int i = tid; i < 32 * H_; i += 128) {
    int k = i / H_, c = i % H_;
    sW1[k][c] = (k < 14) ? W1s[k * H_ + c] : 0.f;
  }
  for (int i = tid; i < H_ * BD_; i += 128) sW2[i / BD_][i % BD_] = W2s[i];
  if (tid < H_)  sb1[tid] = b1s[tid];
  if (tid < BD_) sb2[tid] = b2s[tid];

  if (tid < 64) {
    long r = (long)blockIdx.x * 64 + tid;
    int o = (int)(r % O_);  long t1 = r / O_;
    int n = (int)(t1 % N_); long t2 = t1 / N_;
    int m = (int)(t2 % N_); int b = (int)(t2 / N_);
    const float* pm = pos + (b * N_ + m) * 3;
    const float* pn = pos + (b * N_ + n) * 3;
    float rx = pn[0] - pm[0], ry = pn[1] - pm[1], rz = pn[2] - pm[2];
    const float* oo = ori + o * 3;
    float inv1 = rx * oo[0] + ry * oo[1] + rz * oo[2];
    float nrm  = sqrtf(rx * rx + ry * ry + rz * rz);
    float inv2 = nrm * fabsf(1.0f - inv1);
    float f[14];
    f[0] = inv1; f[1] = inv2;
    f[2] = inv1 * inv1; f[3] = inv1 * inv2; f[4] = inv2 * inv1; f[5] = inv2 * inv2;
#pragma unroll
    for (int i2 = 0; i2 < 4; ++i2) { f[6+2*i2] = f[2+i2]*inv1; f[7+2*i2] = f[2+i2]*inv2; }
#pragma unroll
    for (int k = 0; k < 14; ++k) sF[tid][k] = f[k];
#pragma unroll
    for (int k = 14; k < 32; ++k) sF[tid][k] = 0.f;
  }
  __syncthreads();

  const int w = tid >> 5, lane = tid & 31, half = lane >> 4, mn = lane & 15;
  // GEMM1: (16x32pad) @ (32x128)
#pragma unroll
  for (int ct = 0; ct < 8; ++ct) {
    v8f c = {};
    c = wmma_rm(&sF[w * 16][0], 32, &sW1[0][ct * 16], H_, 32, c);
#pragma unroll
    for (int j = 0; j < 8; ++j) {
      int row = w * 16 + j + half * 8, col = ct * 16 + mn;
      sH1[row][col] = gelu_t(c[j] + sb1[col]);
    }
  }
  __syncthreads();
  // GEMM2: (16x128) @ (128x32)
  long rowbase = (long)blockIdx.x * 64 + w * 16;
#pragma unroll
  for (int ct = 0; ct < 2; ++ct) {
    v8f c = {};
    c = wmma_rm(&sH1[w * 16][0], H_, &sW2[0][ct * 16], BD_, H_, c);
#pragma unroll
    for (int j = 0; j < 8; ++j) {
      int rl = j + half * 8, col = ct * 16 + mn;
      kb[(rowbase + rl) * BD_ + col] = gelu_t(c[j] + sb2[col]);
    }
  }
}

// -------- conv: out[b,m,o,c] = sum_n mask*h[b,n,o,c]*(kb[b,m,n,o,:]@Wsp)[c] --
__global__ void __launch_bounds__(128)
k_conv(int l, const float* Wsp, const float* mask, float* ws) {
  __shared__ float sW[BD_][H_];
  __shared__ float sA[16][BD_];
  __shared__ float sHt[16][H_];
  float* kb = ws + OFF_KB;
  float* h  = ws + OFF_H;
  float* cv = ws + OFF_CONV;
  const int tid = threadIdx.x;
  const int o = blockIdx.x % O_;
  const int m = (blockIdx.x / O_) % N_;
  const int b = blockIdx.x / (O_ * N_);
  for (int i = tid; i < BD_ * H_; i += 128) sW[i / H_][i % H_] = Wsp[(long)l * BD_ * H_ + i];
  const int w = tid >> 5, lane = tid & 31, half = lane >> 4, mn = lane & 15;
  float acc0 = 0.f, acc1 = 0.f;
  for (int n0 = 0; n0 < N_; n0 += 16) {
    __syncthreads();
    for (int i = tid; i < 16 * BD_; i += 128) {
      int r = i / BD_, d = i % BD_;
      sA[r][d] = kb[(((long)(b * N_ + m) * N_ + (n0 + r)) * O_ + o) * BD_ + d];
    }
    for (int i = tid; i < 16 * H_; i += 128) {
      int r = i / H_, c = i % H_;
      sHt[r][c] = h[(((long)(b * N_ + (n0 + r)) * O_) + o) * H_ + c] * mask[b * N_ + n0 + r];
    }
    __syncthreads();
#pragma unroll
    for (int q = 0; q < 2; ++q) {
      int ct = w * 2 + q;
      v8f c = {};
      c = wmma_rm(&sA[0][0], BD_, &sW[0][ct * 16], H_, BD_, c);
      float part = 0.f;
#pragma unroll
      for (int j = 0; j < 8; ++j) part += c[j] * sHt[j + half * 8][ct * 16 + mn];
      if (q == 0) acc0 += part; else acc1 += part;
    }
  }
  acc0 += __shfl_xor(acc0, 16, 32);
  acc1 += __shfl_xor(acc1, 16, 32);
  if (half == 0) {
    long base = ((long)(b * N_ + m) * O_ + o) * H_;
    cv[base + (w * 2 + 0) * 16 + mn] = acc0;
    cv[base + (w * 2 + 1) * 16 + mn] = acc1;
  }
}

// -------- rotation mix + LN + MLP(128->512->128) + residual ----------------
__global__ void __launch_bounds__(128)
k_mix_mlp(int l, const float* conv_b, const float* ln_s, const float* ln_b,
          const float* Wl1, const float* bl1, const float* Wl2, const float* bl2,
          float* ws) {
  __shared__ float sC[O_][H_];
  __shared__ float sA[16][H_];
  __shared__ float sH1[16][WF_ * H_];
  float* h  = ws + OFF_H;
  float* cv = ws + OFF_CONV;
  const float* rot = ws + OFF_ROT + (long)l * O_ * O_ * H_;
  const int tid = threadIdx.x;
  const int m = blockIdx.x % N_, b = blockIdx.x / N_;
  for (int i = tid; i < O_ * H_; i += 128)
    sC[i / H_][i % H_] = cv[((long)(b * N_ + m) * O_) * H_ + i];
  __syncthreads();
  {
    int c = tid;
    for (int p = 0; p < O_; ++p) {
      float acc = 0.f;
      for (int o = 0; o < O_; ++o) acc += sC[o][c] * rot[(p * O_ + o) * H_ + c];
      sA[p][c] = acc * (1.0f / O_) + conv_b[l * H_ + c];
    }
    for (int p = O_; p < 16; ++p) sA[p][c] = 0.f;
  }
  __syncthreads();
  if (tid < O_) {
    float mu = 0.f;
    for (int c = 0; c < H_; ++c) mu += sA[tid][c];
    mu *= (1.0f / H_);
    float var = 0.f;
    for (int c = 0; c < H_; ++c) { float d = sA[tid][c] - mu; var += d * d; }
    var *= (1.0f / H_);
    float rs = rsqrtf(var + 1e-6f);
    for (int c = 0; c < H_; ++c)
      sA[tid][c] = (sA[tid][c] - mu) * rs * ln_s[l * H_ + c] + ln_b[l * H_ + c];
  }
  __syncthreads();
  const int w = tid >> 5, lane = tid & 31, half = lane >> 4, mn = lane & 15;
  // MLP1: (16x128)@(128x512), B read straight from global row-major weights
#pragma unroll
  for (int q = 0; q < 8; ++q) {
    int ct = w * 8 + q;
    v8f c = {};
    c = wmma_rm(&sA[0][0], H_, Wl1 + (long)l * H_ * WF_ * H_ + ct * 16, WF_ * H_, H_, c);
#pragma unroll
    for (int j = 0; j < 8; ++j) {
      int row = j + half * 8, col = ct * 16 + mn;
      sH1[row][col] = gelu_t(c[j] + bl1[l * WF_ * H_ + col]);
    }
  }
  __syncthreads();
  // MLP2: (16x512)@(512x128) + residual
#pragma unroll
  for (int q = 0; q < 2; ++q) {
    int ct = w * 2 + q;
    v8f c = {};
    c = wmma_rm(&sH1[0][0], WF_ * H_, Wl2 + (long)l * WF_ * H_ * H_ + ct * 16, H_, WF_ * H_, c);
#pragma unroll
    for (int j = 0; j < 8; ++j) {
      int p = j + half * 8, col = ct * 16 + mn;
      if (p < O_) {
        long idx = ((long)(b * N_ + m) * O_ + p) * H_ + col;
        h[idx] = c[j] + bl2[l * H_ + col] + h[idx];
      }
    }
  }
}

// -------- masked readout ---------------------------------------------------
__global__ void __launch_bounds__(256)
k_readout(const float* mask, const float* Wro, const float* bro,
          const float* ws, float* out) {
  __shared__ float red[256];
  const int b = blockIdx.x, tid = threadIdx.x;
  const float* h = ws + OFF_H;
  float acc = 0.f;
  for (int idx = tid; idx < N_ * O_; idx += 256) {
    int n = idx / O_, o = idx % O_;
    const float* hr = h + ((long)(b * N_ + n) * O_ + o) * H_;
    float v = bro[0];
    for (int c = 0; c < H_; ++c) v += hr[c] * Wro[c];
    acc += v * mask[b * N_ + n];
  }
  red[tid] = acc;
  __syncthreads();
  for (int s = 128; s > 0; s >>= 1) {
    if (tid < s) red[tid] += red[tid + s];
    __syncthreads();
  }
  if (tid == 0) {
    float msum = 0.f;
    for (int n = 0; n < N_; ++n) msum += mask[b * N_ + n];
    out[b] = red[0] / ((float)O_ * msum);
  }
}

extern "C" void kernel_launch(void* const* d_in, const int* in_sizes, int n_in,
                              void* d_out, int out_size, void* d_ws, size_t ws_size,
                              hipStream_t stream) {
  (void)in_sizes; (void)n_in; (void)out_size; (void)ws_size;
  const float* pos    = (const float*)d_in[0];
  const float* x      = (const float*)d_in[1];
  const float* mask   = (const float*)d_in[2];
  const float* W1s    = (const float*)d_in[3];
  const float* b1s    = (const float*)d_in[4];
  const float* W2s    = (const float*)d_in[5];
  const float* b2s    = (const float*)d_in[6];
  const float* W1r    = (const float*)d_in[7];
  const float* b1r    = (const float*)d_in[8];
  const float* W2r    = (const float*)d_in[9];
  const float* b2r    = (const float*)d_in[10];
  const float* We     = (const float*)d_in[11];
  const float* Wsp    = (const float*)d_in[12];
  const float* Wrot   = (const float*)d_in[13];
  const float* conv_b = (const float*)d_in[14];
  const float* ln_s   = (const float*)d_in[15];
  const float* ln_b   = (const float*)d_in[16];
  const float* Wl1    = (const float*)d_in[17];
  const float* bl1    = (const float*)d_in[18];
  const float* Wl2    = (const float*)d_in[19];
  const float* bl2    = (const float*)d_in[20];
  const float* Wro    = (const float*)d_in[21];
  const float* bro    = (const float*)d_in[22];
  float* ws  = (float*)d_ws;
  float* out = (float*)d_out;

  hipLaunchKernelGGL(k_precompute, dim3(1), dim3(128), 0, stream,
                     W1r, b1r, W2r, b2r, Wrot, ws);
  hipLaunchKernelGGL(k_embed, dim3(B_ * N_), dim3(128), 0, stream, x, We, ws);
  hipLaunchKernelGGL(k_basis, dim3((B_ * N_ * N_ * O_) / 64), dim3(128), 0, stream,
                     pos, W1s, b1s, W2s, b2s, ws);
  for (int l = 0; l < L_; ++l) {
    hipLaunchKernelGGL(k_conv, dim3(B_ * N_ * O_), dim3(128), 0, stream, l, Wsp, mask, ws);
    hipLaunchKernelGGL(k_mix_mlp, dim3(B_ * N_), dim3(128), 0, stream,
                       l, conv_b, ln_s, ln_b, Wl1, bl1, Wl2, bl2, ws);
  }
  hipLaunchKernelGGL(k_readout, dim3(B_), dim3(256), 0, stream, mask, Wro, bro, ws, out);
}
